// networkarch_12146167513772
// MI455X (gfx1250) — matrix operand order repaired and checked by reading, not compiled, not run on hardware
//
#include <hip/hip_runtime.h>

typedef __attribute__((ext_vector_type(16))) _Float16 v16h;
typedef __attribute__((ext_vector_type(8)))  _Float16 v8h;
typedef __attribute__((ext_vector_type(8)))  float    v8f;

// ---------------- problem dims ----------------
#define BATCH   2048
#define LEN_T   51
#define N_SHIFT 50
#define M_ROWS  (BATCH*LEN_T)        // 104448 rows (16-row tiles: 6528)
#define MTILES  (M_ROWS/16)          // 6528
#define WAVES   8                    // 8 waves/block -> 2 per SIMD32
#define THREADS (WAVES*32)

// ---------------- workspace layout (units: _Float16) ----------------
#define WS_ENC_WT 0                  // 5 x 128x128 transposed f16 (enc_W[0..3], enc_Wout)
#define WS_DEC_WT 81920              // 5 x 128x128 transposed f16
#define WS_WEI_T  163840             // W_enc_inner^T padded [32][128]
#define WS_WDI_T  167936             // W_dec_inner^T padded [128][32]
#define WS_LPOW   172032             // 50 x [32][32] transposed f16 (L^1..L^50)
#define WS_FET0   223232             // full_enc at t=0: [2048][32] f16
#define WS_PENC   288768             // part_enc f16 [104448][128]
#define WS_PDEC   13658112           // part_dec f16 [104448][128]

// ---------------- output layout (floats) ----------------
#define OUT_AUTO  0
#define OUT_OUTER 13369344
#define OUT_PRED  26738688

// ---------------- shared mem byte sizes ----------------
#define SM_ENC  (182784 + WAVES*9216)   // 256512
#define SM_DEC  (166400 + WAVES*8192)   // 231936
#define SM_PRED (174592 + WAVES*9216)   // 248320

// =====================================================================
// TDM weight staging (CDNA5 Tensor Data Mover). 1-D descriptor:
// group0 = {count=1, lds_addr, global_addr[56:0], type=2}
// group1 = {data_size=8B, tensor_dim0=tile_dim0=stride=nbytes/8, dim1=1}
// =====================================================================
#if defined(__HIP_DEVICE_COMPILE__) && __has_builtin(__builtin_amdgcn_tensor_load_to_lds)
#define HAVE_TDM 1
typedef __attribute__((ext_vector_type(4))) unsigned int tdm_u32x4;
typedef __attribute__((ext_vector_type(8))) int          tdm_i32x8;
typedef __attribute__((ext_vector_type(4))) int          tdm_i32x4;

__device__ __forceinline__
void tdm_load_to_lds(void* lds_dst, const void* gsrc, unsigned nbytes) {
  unsigned lds_off = (unsigned)(unsigned long long)
      (__attribute__((address_space(3))) const void*)lds_dst;
  unsigned long long ga = (unsigned long long)gsrc;
  unsigned units = nbytes >> 3;                   // 8-byte elements, <= 65535
  tdm_u32x4 g0;
  g0[0] = 1u;                                     // count=1, user mode
  g0[1] = lds_off;                                // lds_addr (bytes)
  g0[2] = (unsigned)ga;                           // global_addr[31:0]
  g0[3] = (unsigned)((ga >> 32) & 0x01FFFFFFu) | 0x80000000u; // ga[56:32] | type=2
  tdm_i32x8 g1;
  g1[0] = (int)(3u << 16);                        // data_size = 8B
  g1[1] = (int)((units & 0xFFFFu) << 16);         // tensor_dim0[15:0]
  g1[2] = (int)((units >> 16) | (1u << 16));      // tensor_dim0[31:16] | tensor_dim1=1
  g1[3] = (int)((units & 0xFFFFu) << 16);         // tile_dim0
  g1[4] = 0;                                      // tile_dim1=0 (1-D), tile_dim2=0
  g1[5] = (int)units;                             // tensor_dim0_stride
  g1[6] = 0;
  g1[7] = 0;
  tdm_i32x4 z4 = {0, 0, 0, 0};
#if __clang_major__ >= 23
  tdm_i32x8 z8 = {0, 0, 0, 0, 0, 0, 0, 0};
  __builtin_amdgcn_tensor_load_to_lds(g0, g1, z4, z4, z8, 0);
#else
  __builtin_amdgcn_tensor_load_to_lds(g0, g1, z4, z4, 0);
#endif
}
#endif

__device__ __forceinline__
void cpy16(uint4* d, const uint4* s, int n, int tid, int nt) {
  for (int i = tid; i < n; i += nt) d[i] = s[i];
}

// stage a weight panel: TDM DMA (wave 0 issues) or vector-copy fallback
__device__ __forceinline__
void stage_panel(void* lds_dst, const void* gsrc, unsigned nbytes,
                 int tid, int wave) {
#if defined(HAVE_TDM)
  if (wave == 0) tdm_load_to_lds(lds_dst, gsrc, nbytes);
  (void)tid;
#else
  cpy16((uint4*)lds_dst, (const uint4*)gsrc, (int)(nbytes >> 4), tid, THREADS);
#endif
}

__device__ __forceinline__ void stage_fence(int wave) {
#if defined(HAVE_TDM)
  if (wave == 0) __builtin_amdgcn_s_wait_tensorcnt(0);
#else
  (void)wave;
#endif
  __syncthreads();
}

// =====================================================================
// Fragment loaders.
// A (16x32 f16): lane l<16 holds row l, K {0..7,16..23}; lane l>=16 row l-16,
// K {8..15,24..31}  -> two 16B chunks from row-major LDS.
// B (32x16 f16): lane l holds col (l&15); same K split as A
// -> two 16B chunks per lane from *transposed* [N][K] weights.
// =====================================================================
__device__ __forceinline__ v16h load_afrag(const _Float16* buf, int stride,
                                           int kstep, int l16, int half) {
  const _Float16* p = buf + l16 * stride + kstep * 32 + half * 8;
  v8h lo = *(const v8h*)p;
  v8h hi = *(const v8h*)(p + 16);
  return __builtin_shufflevector(lo, hi, 0,1,2,3,4,5,6,7,8,9,10,11,12,13,14,15);
}

__device__ __forceinline__ v16h load_bfrag(const _Float16* wT, int wStride,
                                           int ntile, int kstep, int l16, int half) {
  const _Float16* p = wT + (ntile * 16 + l16) * wStride + kstep * 32 + half * 8;
  v8h lo = *(const v8h*)p;
  v8h hi = *(const v8h*)(p + 16);
  return __builtin_shufflevector(lo, hi, 0,1,2,3,4,5,6,7,8,9,10,11,12,13,14,15);
}

// Generic 16xN layer: out = act(A[16,K] * W^T + bias (+skip)), f16 result to LDS.
template<int KSTEPS, int NTILES, bool RELU, bool SKIP, bool HASBIAS>
__device__ __forceinline__
void mm_layer(const _Float16* __restrict__ aBuf, int aStride,
              const _Float16* __restrict__ wT,  int wStride,
              const float*    __restrict__ bias,
              const _Float16* __restrict__ skipBuf, int skipStride,
              _Float16*       __restrict__ outBuf,  int outStride,
              int l16, int half) {
  v16h afrag[KSTEPS];
#pragma unroll
  for (int k = 0; k < KSTEPS; ++k) afrag[k] = load_afrag(aBuf, aStride, k, l16, half);
#pragma unroll
  for (int nt = 0; nt < NTILES; ++nt) {
    v8f acc;
    float bv = 0.0f;
    if constexpr (HASBIAS) bv = bias[nt * 16 + l16];
#pragma unroll
    for (int r = 0; r < 8; ++r) acc[r] = bv;
#pragma unroll
    for (int k = 0; k < KSTEPS; ++k) {
      v16h bfrag = load_bfrag(wT, wStride, nt, k, l16, half);
      acc = __builtin_amdgcn_wmma_f32_16x16x32_f16(false, afrag[k], false, bfrag,
                                                   (short)0, acc, false, false);
    }
    int ncol = nt * 16 + l16;
#pragma unroll
    for (int r = 0; r < 8; ++r) {
      float v = acc[r];
      if constexpr (SKIP) v += (float)skipBuf[(r + half * 8) * skipStride + ncol];
      if constexpr (RELU) v = fmaxf(v, 0.0f);
      outBuf[(r + half * 8) * outStride + ncol] = (_Float16)v;
    }
  }
}

// Final layer of a res_block: K=128, N=128, +bias +skip, f32 result to global.
__device__ __forceinline__
void mm_layer_out(const _Float16* __restrict__ aBuf,
                  const _Float16* __restrict__ wT,
                  const float* __restrict__ bias,
                  const _Float16* __restrict__ skipBuf,
                  float* __restrict__ gBase, int gStride,
                  int l16, int half) {
  v16h afrag[4];
#pragma unroll
  for (int k = 0; k < 4; ++k) afrag[k] = load_afrag(aBuf, 128, k, l16, half);
#pragma unroll
  for (int nt = 0; nt < 8; ++nt) {
    v8f acc;
    float bv = bias[nt * 16 + l16];
#pragma unroll
    for (int r = 0; r < 8; ++r) acc[r] = bv;
#pragma unroll
    for (int k = 0; k < 4; ++k) {
      v16h bfrag = load_bfrag(wT, 128, nt, k, l16, half);
      acc = __builtin_amdgcn_wmma_f32_16x16x32_f16(false, afrag[k], false, bfrag,
                                                   (short)0, acc, false, false);
    }
    int ncol = nt * 16 + l16;
#pragma unroll
    for (int r = 0; r < 8; ++r) {
      int row = r + half * 8;
      gBase[row * gStride + ncol] = acc[r] + (float)skipBuf[row * 128 + ncol];
    }
  }
}

// dec res_block: 4 ReLU layers + output layer with identity skip -> global f32
__device__ __forceinline__
void dec_chain(const _Float16* inbuf, _Float16* actbuf,
               const _Float16* decWT, const float* bias,
               float* gBase, int gStride, int l16, int half) {
  mm_layer<4,8,true,false,true>(inbuf,128,  decWT+0*16384,128, bias+0,   nullptr,128, actbuf,128, l16,half);
  mm_layer<4,8,true,false,true>(actbuf,128, decWT+1*16384,128, bias+128, nullptr,128, actbuf,128, l16,half);
  mm_layer<4,8,true,false,true>(actbuf,128, decWT+2*16384,128, bias+256, nullptr,128, actbuf,128, l16,half);
  mm_layer<4,8,true,false,true>(actbuf,128, decWT+3*16384,128, bias+384, nullptr,128, actbuf,128, l16,half);
  mm_layer_out(actbuf, decWT+4*16384, bias+512, inbuf, gBase, gStride, l16, half);
}

// =====================================================================
// Prep: transpose + f32->f16 all weight matrices into workspace.
// =====================================================================
__global__ void k_prep(const float* __restrict__ encW, const float* __restrict__ encWout,
                       const float* __restrict__ decW, const float* __restrict__ decWout,
                       const float* __restrict__ wei,  const float* __restrict__ wdi,
                       _Float16* __restrict__ ws) {
  int b = blockIdx.x, t = threadIdx.x;
  if (b < 10) {
    const float* src; _Float16* dst;
    if (b < 4)      { src = encW + b * 16384;      dst = ws + WS_ENC_WT + b * 16384; }
    else if (b == 4){ src = encWout;               dst = ws + WS_ENC_WT + 4 * 16384; }
    else if (b < 9) { src = decW + (b - 5) * 16384; dst = ws + WS_DEC_WT + (b - 5) * 16384; }
    else            { src = decWout;               dst = ws + WS_DEC_WT + 4 * 16384; }
    for (int i = t; i < 16384; i += 256) {
      int n = i >> 7, k = i & 127;
      dst[i] = (_Float16)src[k * 128 + n];
    }
  } else if (b == 10) {            // W_enc_inner [128,21] -> [32][128] zero-padded
    _Float16* dst = ws + WS_WEI_T;
    for (int i = t; i < 4096; i += 256) {
      int n = i >> 7, k = i & 127;
      dst[i] = (n < 21) ? (_Float16)wei[k * 21 + n] : (_Float16)0.0f;
    }
  } else {                         // W_dec_inner [21,128] -> [128][32] zero-padded
    _Float16* dst = ws + WS_WDI_T;
    for (int i = t; i < 4096; i += 256) {
      int n = i >> 5, k = i & 31;
      dst[i] = (k < 21) ? (_Float16)wdi[k * 128 + n] : (_Float16)0.0f;
    }
  }
}

// =====================================================================
// Koopman powers: L^1..L^50 (f32 chain, f16 transposed snapshots).
// Turns the sequential scan into 2048x50 independent WMMA tasks.
// =====================================================================
__global__ __launch_bounds__(1024) void k_lpow(const float* __restrict__ L,
                                               _Float16* __restrict__ lpow) {
  __shared__ float P[1024], Lp[1024];
  int t = threadIdx.x, i = t >> 5, j = t & 31;
  float lv = (i < 21 && j < 21) ? L[i * 21 + j] : 0.0f;
  Lp[t] = lv; P[t] = lv;
  __syncthreads();
  for (int s = 0; s < 50; ++s) {
    lpow[s * 1024 + j * 32 + i] = (_Float16)P[i * 32 + j];  // store P^T (B layout)
    float acc = 0.0f;
    for (int m = 0; m < 32; ++m) acc += P[i * 32 + m] * Lp[m * 32 + j];
    __syncthreads();
    P[t] = acc;
    __syncthreads();
  }
}

// =====================================================================
// Encoder: enc res_block + 128->21->128 bottleneck on all 104448 rows.
// Emits part_enc / part_dec (f16, ws) and full_enc rows at t==0.
// =====================================================================
__global__ __launch_bounds__(THREADS)
void k_enc(const float* __restrict__ inputs,
           const float* __restrict__ enc_b, const float* __restrict__ enc_bout,
           _Float16* __restrict__ ws) {
  extern __shared__ __align__(16) unsigned char smem[];
  _Float16* sEncWT = (_Float16*)smem;                 // 163840 B
  _Float16* sWeiT  = (_Float16*)(smem + 163840);      // 8192 B
  _Float16* sWdiT  = (_Float16*)(smem + 172032);      // 8192 B
  float*    sBias  = (float*)(smem + 180224);         // 640 f32

  int tid = threadIdx.x, wave = tid >> 5, lane = tid & 31;
  int l16 = lane & 15, half = lane >> 4;
  unsigned char* wbase = smem + 182784 + wave * 9216;
  _Float16* xbuf   = (_Float16*)wbase;                // 16x128 f16
  _Float16* actbuf = (_Float16*)(wbase + 4096);       // 16x128 f16
  _Float16* febuf  = (_Float16*)(wbase + 8192);       // 16x32  f16

  stage_panel(sEncWT, ws + WS_ENC_WT, 163840, tid, wave);
  stage_panel(sWeiT,  ws + WS_WEI_T,  8192,   tid, wave);
  stage_panel(sWdiT,  ws + WS_WDI_T,  8192,   tid, wave);
  for (int i = tid; i < 512; i += THREADS) sBias[i] = enc_b[i];
  for (int i = tid; i < 128; i += THREADS) sBias[512 + i] = enc_bout[i];
  stage_fence(wave);

  int r0 = (blockIdx.x * WAVES + wave) * 16;

  // load x tile (f32 -> f16, coalesced)
  const float* src = inputs + (size_t)r0 * 128;
  for (int i = 0; i < 16; ++i) {
    int e = (i * 32 + lane) * 4;
    float4 v = *(const float4*)(src + e);
    _Float16* d = xbuf + e;
    d[0] = (_Float16)v.x; d[1] = (_Float16)v.y;
    d[2] = (_Float16)v.z; d[3] = (_Float16)v.w;
  }

  // enc res_block
  mm_layer<4,8,true,false,true>(xbuf,128,  sEncWT+0*16384,128, sBias+0,   nullptr,128, actbuf,128, l16,half);
  mm_layer<4,8,true,false,true>(actbuf,128, sEncWT+1*16384,128, sBias+128, nullptr,128, actbuf,128, l16,half);
  mm_layer<4,8,true,false,true>(actbuf,128, sEncWT+2*16384,128, sBias+256, nullptr,128, actbuf,128, l16,half);
  mm_layer<4,8,true,false,true>(actbuf,128, sEncWT+3*16384,128, sBias+384, nullptr,128, actbuf,128, l16,half);
  mm_layer<4,8,false,true,true>(actbuf,128, sEncWT+4*16384,128, sBias+512, xbuf,128,    actbuf,128, l16,half);

  // part_enc -> workspace (f16, coalesced)
  {
    uint4* pe = (uint4*)(ws + WS_PENC) + r0 * 16;
    const uint4* sa = (const uint4*)actbuf;
    for (int i = lane; i < 256; i += 32) pe[i] = sa[i];
  }

  // full_enc = part_enc @ W_enc_inner (padded N=32)
  mm_layer<4,2,false,false,false>(actbuf,128, sWeiT,128, nullptr, nullptr,128, febuf,32, l16,half);

  // full_enc rows at t==0 -> workspace (feeds prediction kernel)
  for (int idx = lane; idx < 512; idx += 32) {
    int m = idx >> 5, n = idx & 31;
    int row = r0 + m;
    if (row % LEN_T == 0) ws[WS_FET0 + (row / LEN_T) * 32 + n] = febuf[m * 32 + n];
  }

  // part_dec = full_enc @ W_dec_inner (K padded to 32)
  mm_layer<1,8,false,false,false>(febuf,32, sWdiT,32, nullptr, nullptr,128, actbuf,128, l16,half);
  {
    uint4* pd = (uint4*)(ws + WS_PDEC) + r0 * 16;
    const uint4* sa = (const uint4*)actbuf;
    for (int i = lane; i < 256; i += 32) pd[i] = sa[i];
  }
}

// =====================================================================
// Decoder: dec res_block on part_dec -> autoencoder_output, and on
// part_enc -> outer_auto_output. Shares dec weights in LDS.
// =====================================================================
__global__ __launch_bounds__(THREADS)
void k_dec(const _Float16* __restrict__ ws,
           const float* __restrict__ dec_b, const float* __restrict__ dec_bout,
           float* __restrict__ out) {
  extern __shared__ __align__(16) unsigned char smem[];
  _Float16* sDecWT = (_Float16*)smem;                 // 163840 B
  float*    sBias  = (float*)(smem + 163840);         // 640 f32

  int tid = threadIdx.x, wave = tid >> 5, lane = tid & 31;
  int l16 = lane & 15, half = lane >> 4;
  unsigned char* wbase = smem + 166400 + wave * 8192;
  _Float16* inbuf  = (_Float16*)wbase;
  _Float16* actbuf = (_Float16*)(wbase + 4096);

  stage_panel(sDecWT, ws + WS_DEC_WT, 163840, tid, wave);
  for (int i = tid; i < 512; i += THREADS) sBias[i] = dec_b[i];
  for (int i = tid; i < 128; i += THREADS) sBias[512 + i] = dec_bout[i];
  stage_fence(wave);

  int r0 = (blockIdx.x * WAVES + wave) * 16;

  // chain 1: part_dec -> autoencoder_output
  {
    const uint4* ps = (const uint4*)(ws + WS_PDEC) + r0 * 16;
    uint4* di = (uint4*)inbuf;
    for (int i = lane; i < 256; i += 32) di[i] = ps[i];
  }
  dec_chain(inbuf, actbuf, sDecWT, sBias, out + OUT_AUTO + (size_t)r0 * 128, 128, l16, half);

  // chain 2: part_enc -> outer_auto_output
  {
    const uint4* ps = (const uint4*)(ws + WS_PENC) + r0 * 16;
    uint4* di = (uint4*)inbuf;
    for (int i = lane; i < 256; i += 32) di[i] = ps[i];
  }
  dec_chain(inbuf, actbuf, sDecWT, sBias, out + OUT_OUTER + (size_t)r0 * 128, 128, l16, half);
}

// =====================================================================
// Prediction: one (16-row batch tile, shift s) task per wave.
// adv = full_enc @ L^{s+1};  pred = dec_res_block(adv @ W_dec_inner)
// =====================================================================
__global__ __launch_bounds__(THREADS)
void k_pred(const _Float16* __restrict__ ws,
            const float* __restrict__ dec_b, const float* __restrict__ dec_bout,
            float* __restrict__ out) {
  extern __shared__ __align__(16) unsigned char smem[];
  _Float16* sDecWT = (_Float16*)smem;                 // 163840 B
  _Float16* sWdiT  = (_Float16*)(smem + 163840);      // 8192 B
  float*    sBias  = (float*)(smem + 172032);         // 640 f32

  int tid = threadIdx.x, wave = tid >> 5, lane = tid & 31;
  int l16 = lane & 15, half = lane >> 4;
  unsigned char* wbase = smem + 174592 + wave * 9216;
  _Float16* febuf  = (_Float16*)wbase;                // 16x32
  _Float16* inbuf  = (_Float16*)(wbase + 1024);       // 16x128
  _Float16* actbuf = (_Float16*)(wbase + 5120);       // 16x128

  stage_panel(sDecWT, ws + WS_DEC_WT, 163840, tid, wave);
  stage_panel(sWdiT,  ws + WS_WDI_T,  8192,   tid, wave);
  for (int i = tid; i < 512; i += THREADS) sBias[i] = dec_b[i];
  for (int i = tid; i < 128; i += THREADS) sBias[512 + i] = dec_bout[i];
  stage_fence(wave);

  int task = blockIdx.x * WAVES + wave;   // 6400 tasks = 128 tiles x 50 shifts
  int mtile = task / N_SHIFT;
  int s     = task % N_SHIFT;

  {
    const uint4* fs = (const uint4*)(ws + WS_FET0) + mtile * 64;
    uint4* df = (uint4*)febuf;
    for (int i = lane; i < 64; i += 32) df[i] = fs[i];
  }

  // adv = fe @ (L^{s+1})   (B frags streamed from L2-resident workspace)
  const _Float16* lp = ws + WS_LPOW + s * 1024;
  mm_layer<1,2,false,false,false>(febuf,32, lp,32, nullptr, nullptr,32, febuf,32, l16,half);

  // tmp = adv @ W_dec_inner -> res_block input
  mm_layer<1,8,false,false,false>(febuf,32, sWdiT,32, nullptr, nullptr,128, inbuf,128, l16,half);

  float* gBase = out + OUT_PRED + (size_t)(mtile * 16) * (N_SHIFT * 128) + s * 128;
  dec_chain(inbuf, actbuf, sDecWT, sBias, gBase, N_SHIFT * 128, l16, half);
}

// =====================================================================
extern "C" void kernel_launch(void* const* d_in, const int* in_sizes, int n_in,
                              void* d_out, int out_size, void* d_ws, size_t ws_size,
                              hipStream_t stream) {
  (void)in_sizes; (void)n_in; (void)out_size; (void)ws_size;
  const float* inputs   = (const float*)d_in[0];
  const float* enc_W    = (const float*)d_in[1];
  const float* enc_b    = (const float*)d_in[2];
  const float* enc_Wout = (const float*)d_in[3];
  const float* enc_bout = (const float*)d_in[4];
  const float* dec_W    = (const float*)d_in[5];
  const float* dec_b    = (const float*)d_in[6];
  const float* dec_Wout = (const float*)d_in[7];
  const float* dec_bout = (const float*)d_in[8];
  const float* wei      = (const float*)d_in[9];
  const float* L        = (const float*)d_in[10];
  const float* wdi      = (const float*)d_in[11];
  _Float16* ws  = (_Float16*)d_ws;
  float*    out = (float*)d_out;

  (void)hipFuncSetAttribute((const void*)k_enc,  hipFuncAttributeMaxDynamicSharedMemorySize, SM_ENC);
  (void)hipFuncSetAttribute((const void*)k_dec,  hipFuncAttributeMaxDynamicSharedMemorySize, SM_DEC);
  (void)hipFuncSetAttribute((const void*)k_pred, hipFuncAttributeMaxDynamicSharedMemorySize, SM_PRED);

  k_prep<<<12, 256, 0, stream>>>(enc_W, enc_Wout, dec_W, dec_Wout, wei, wdi, ws);
  k_lpow<<<1, 1024, 0, stream>>>(L, ws + WS_LPOW);
  k_enc<<<MTILES / WAVES, THREADS, SM_ENC, stream>>>(inputs, enc_b, enc_bout, ws);
  k_dec<<<MTILES / WAVES, THREADS, SM_DEC, stream>>>(ws, dec_b, dec_bout, out);
  k_pred<<<(BATCH / 16) * N_SHIFT / WAVES, THREADS, SM_PRED, stream>>>(ws, dec_b, dec_bout, out);
}
